// Seq2Seq_20727512170627
// MI455X (gfx1250) — compile-verified, matrix operand order checked by reading, not run on hardware
//
#include <hip/hip_runtime.h>
#include <hip/hip_bf16.h>

typedef __bf16 bf16;
typedef bf16  v16bf __attribute__((ext_vector_type(16)));
typedef float v8f   __attribute__((ext_vector_type(8)));

#define B_    128
#define LINS_ 64
#define LHIS_ 128
#define LACT_ 32
#define ACTSZ_ 100
#define INS_H_ 256
#define INS_E_ 128
#define ACT_E_ 128
#define ACT_H_ 512
#define ENVD_  448   // 7*(POS_E+COL_H)

// ---------------------------------------------------------------------------
// WMMA helpers (CDNA5 v_wmma_f32_16x16x32_bf16, wave32)
// A/B fragment: row r = lane&15 of a row-major [16 x 32] slab (ld elements/row),
// K pairs per VGPR per ISA 7.12.2; lanes 16-31 take K+8 within each 16-block.
// ---------------------------------------------------------------------------
__device__ __forceinline__ int lane_id_() { return threadIdx.x & 31; }

__device__ __forceinline__ v16bf frag_ld(const bf16* __restrict__ p, int ld) {
  const int l = lane_id_();
  const bf16* row = p + (l & 15) * ld + ((l >> 4) & 1) * 8;
  v16bf f;
#pragma unroll
  for (int i = 0; i < 8; ++i) {
    const int kb = (i < 4) ? (2 * i) : (8 + 2 * i);   // {0,2,4,6,16,18,20,22}
    f[2 * i]     = row[kb];
    f[2 * i + 1] = row[kb + 1];
  }
  return f;
}

__device__ __forceinline__ v8f wmma_bf(v16bf a, v16bf b, v8f c) {
  return __builtin_amdgcn_wmma_f32_16x16x32_bf16(false, a, false, b, (short)0, c,
                                                 false, false);
}

__device__ __forceinline__ float sigf(float x) { return 1.0f / (1.0f + __expf(-x)); }

// ---------------------------------------------------------------------------
// Generic tiled GEMM: out[M,N] = A[M,K](bf16) * W^T  (W stored [N,K] bf16)
//   optional bias[N] (f32), optional f32 C-init (row % cmod), f32/bf16 outputs.
// One 16x16 tile per wave; blockDim = 256 (8 waves).
// ---------------------------------------------------------------------------
__global__ void k_gemm(const bf16* __restrict__ A, int lda,
                       const bf16* __restrict__ W, int ldw,
                       const float* __restrict__ bias,
                       const float* __restrict__ cinit, int cmod, int cld,
                       float* outf, bf16* outbf, int old_,
                       int M, int Npad, int Nreal, int K) {
  const int wavesPerBlk = blockDim.x >> 5;
  const int wave = blockIdx.x * wavesPerBlk + (threadIdx.x >> 5);
  const int ntn = Npad >> 4;
  const int total = (M >> 4) * ntn;
  if (wave >= total) return;                       // wave-uniform exit
  const int mt = wave / ntn, nt = wave % ntn;
  const int l = lane_id_(), n = l & 15, mb = (l >> 4) * 8;

  v8f acc = {0.f, 0.f, 0.f, 0.f, 0.f, 0.f, 0.f, 0.f};
  if (bias) {
    const float bv = bias[nt * 16 + n];
#pragma unroll
    for (int e = 0; e < 8; ++e) acc[e] = bv;
  }
  if (cinit) {
#pragma unroll
    for (int e = 0; e < 8; ++e) {
      int row = mt * 16 + mb + e;
      if (cmod) row %= cmod;
      acc[e] += cinit[(size_t)row * cld + nt * 16 + n];
    }
  }
  const bf16* Ab = A + (size_t)(mt * 16) * lda;
  const bf16* Wb = W + (size_t)(nt * 16) * ldw;
  for (int kb = 0; kb < K; kb += 32) {
    __builtin_prefetch((const void*)(Wb + kb + 64), 0, 1);
    v16bf af = frag_ld(Ab + kb, lda);
    v16bf bfr = frag_ld(Wb + kb, ldw);
    acc = wmma_bf(af, bfr, acc);
  }
  const int col = nt * 16 + n;
  if (col < Nreal) {
#pragma unroll
    for (int e = 0; e < 8; ++e) {
      const int row = mt * 16 + mb + e;
      if (outf)  outf[(size_t)row * old_ + col] = acc[e];
      if (outbf) outbf[(size_t)row * old_ + col] = (bf16)acc[e];
    }
  }
}

// ---------------------------------------------------------------------------
// Conversion / gather / small elementwise kernels
// ---------------------------------------------------------------------------
__global__ void k_conv(const float* __restrict__ s, bf16* __restrict__ d,
                       int rows, int cols, int dld, int coff) {
  const int i = blockIdx.x * blockDim.x + threadIdx.x;
  if (i >= rows * cols) return;
  const int r = i / cols, c = i % cols;
  d[(size_t)r * dld + coff + c] = (bf16)s[i];
}

__global__ void k_convT(const float* __restrict__ s, bf16* __restrict__ d,
                        int K, int N) {                 // s[K,N] -> d[N,K]
  const int i = blockIdx.x * blockDim.x + threadIdx.x;
  if (i >= K * N) return;
  const int k = i / N, n = i % N;
  d[(size_t)n * K + k] = (bf16)s[i];
}

__global__ void k_addv(const float* a, const float* b, float* o, int n) {
  const int i = blockIdx.x * blockDim.x + threadIdx.x;
  if (i < n) o[i] = a[i] + b[i];
}

__global__ void k_pad_h2a(const float* __restrict__ w, const float* __restrict__ b,
                          bf16* __restrict__ wo, float* __restrict__ bo) {
  const int i = blockIdx.x * blockDim.x + threadIdx.x;   // 112*512
  if (i < 112 * 512) {
    const int r = i / 512, c = i % 512;
    wo[i] = (bf16)((r < ACTSZ_) ? w[r * 512 + c] : 0.0f);
  }
  if (i < 112) bo[i] = (i < ACTSZ_) ? b[i] : 0.0f;
}

__global__ void k_gather_tok(const int* __restrict__ tok, const float* __restrict__ emb,
                             bf16* __restrict__ out, int Bn, int L) {
  const int i = blockIdx.x * blockDim.x + threadIdx.x;   // (t*Bn+b)*128+e
  if (i >= Bn * L * INS_E_) return;
  const int e = i % INS_E_;
  const int tb = i / INS_E_;
  const int b = tb % Bn, t = tb / Bn;
  out[i] = (bf16)emb[(size_t)tok[b * L + t] * INS_E_ + e];
}

__global__ void k_gather_env(const int* __restrict__ env, const float* __restrict__ cemb,
                             bf16* __restrict__ out) {   // (s*896 + b*7+j)*32+e
  const int i = blockIdx.x * blockDim.x + threadIdx.x;
  if (i >= 4 * 896 * 32) return;
  const int e = i & 31;
  const int rs = i >> 5;
  const int r = rs % 896, s = rs / 896;
  const int b = r / 7, j = r % 7;
  out[i] = (bf16)cemb[(size_t)env[b * 35 + j * 5 + 1 + s] * 32 + e];
}

__global__ void k_gather_act(const int* __restrict__ act, const float* __restrict__ emb,
                             bf16* __restrict__ out) {   // (t*128+b)*128+e
  const int i = blockIdx.x * blockDim.x + threadIdx.x;
  if (i >= LACT_ * B_ * ACT_E_) return;
  const int e = i % ACT_E_;
  const int tb = i / ACT_E_;
  const int b = tb % B_, t = tb / B_;
  out[i] = (bf16)emb[(size_t)act[b * LACT_ + t] * ACT_E_ + e];
}

// env context: ctx[b][j*64+u] = u<32 ? pos[j][u] : h_last[b*7+j][u-32]
// written twice into envcat (the broadcast attends collapse to identity)
__global__ void k_ctx(const float* __restrict__ hslast, const float* __restrict__ pos,
                      bf16* __restrict__ envcat, int o1, int o2) {
  const int i = blockIdx.x * blockDim.x + threadIdx.x;   // b*448 + ju
  if (i >= B_ * ENVD_) return;
  const int b = i / ENVD_, ju = i % ENVD_;
  const int j = ju >> 6, u = ju & 63;
  const float v = (u < 32) ? pos[j * 32 + u] : hslast[(b * 7 + j) * 32 + (u - 32)];
  envcat[(size_t)b * 1792 + o1 + ju] = (bf16)v;
  envcat[(size_t)b * 1792 + o2 + ju] = (bf16)v;
}

// ---------------------------------------------------------------------------
// Persistent LSTM scan. One block per run; h double-buffered in LDS (bf16),
// c in registers. Per step: gates = [x|h] @ [Wih|Whh]^T + b via WMMA.
// ---------------------------------------------------------------------------
struct LstmCfg {
  const bf16* xbf;     // [T*Bn, E] rows t*Bn+b
  const bf16* wcomb;   // [4H, E+H] bf16
  const float* bias;   // [4H] = bih+bhh
  float* outf;         // may be null
  bf16* outbf;         // may be null
  int T, Bn, H, E, ob, ot, coff, rev;
};
struct Lstm6 { LstmCfg c[6]; };

__global__ void k_lstm(Lstm6 all) {
  extern __shared__ bf16 hbuf[];                 // 2 * 32768 bf16
  const LstmCfg cfg = all.c[blockIdx.x];
  const int BH = cfg.Bn * cfg.H;
  const int tid = threadIdx.x;
  for (int i = tid; i < BH; i += blockDim.x) hbuf[i] = (bf16)0.0f;
  __syncthreads();

  const int wave = tid >> 5, nw = blockDim.x >> 5;
  const int l = tid & 31, n = l & 15, mb = (l >> 4) * 8;
  const int Hn = cfg.H >> 4;
  const int GB = (cfg.Bn >> 4) * Hn;
  const int EH = cfg.E + cfg.H;
  const v8f vzero = {0.f, 0.f, 0.f, 0.f, 0.f, 0.f, 0.f, 0.f};
  v8f carr[4];
#pragma unroll
  for (int q = 0; q < 4; ++q) carr[q] = vzero;

  int cur = 0;
  for (int t = 0; t < cfg.T; ++t) {
    const int tx = cfg.rev ? (cfg.T - 1 - t) : t;  // input & output index
    const bf16* xrow = cfg.xbf + (size_t)(tx * cfg.Bn) * cfg.E;
    const bf16* hprev = hbuf + cur * 32768;
    bf16* hnext = hbuf + (cur ^ 1) * 32768;
    int idx = 0;
    for (int gb = wave; gb < GB; gb += nw, ++idx) {
      const int mt = gb / Hn, nt = gb % Hn;
      v8f g[4];
#pragma unroll
      for (int gi = 0; gi < 4; ++gi) {
        const int ngl = gi * cfg.H + nt * 16;
        const float bv = cfg.bias[ngl + n];
        v8f acc;
#pragma unroll
        for (int e = 0; e < 8; ++e) acc[e] = bv;
        const bf16* wr = cfg.wcomb + (size_t)ngl * EH;
        for (int kb = 0; kb < cfg.E; kb += 32)
          acc = wmma_bf(frag_ld(xrow + (size_t)mt * 16 * cfg.E + kb, cfg.E),
                        frag_ld(wr + kb, EH), acc);
        for (int kb = 0; kb < cfg.H; kb += 32)
          acc = wmma_bf(frag_ld(hprev + (size_t)mt * 16 * cfg.H + kb, cfg.H),
                        frag_ld(wr + cfg.E + kb, EH), acc);
        g[gi] = acc;
      }
#pragma unroll
      for (int e = 0; e < 8; ++e) {
        const float iv = sigf(g[0][e]);
        const float fv = sigf(g[1][e]);
        const float gv = tanhf(g[2][e]);
        const float ov = sigf(g[3][e]);
        const float cv = fv * carr[idx][e] + iv * gv;
        carr[idx][e] = cv;
        const float hv = ov * tanhf(cv);
        const int brow = mt * 16 + mb + e;
        const int col = nt * 16 + n;
        hnext[brow * cfg.H + col] = (bf16)hv;
        const size_t oi = (size_t)brow * cfg.ob + (size_t)tx * cfg.ot + cfg.coff + col;
        if (cfg.outf)  cfg.outf[oi] = hv;
        if (cfg.outbf) cfg.outbf[oi] = (bf16)hv;
      }
    }
    __syncthreads();
    cur ^= 1;
  }
}

// ---------------------------------------------------------------------------
// Persistent decoder. h double-buffered in LDS (bf16, 2*128*512), c in regs.
// Attention phases in VALU, hk + LSTM-gate GEMMs in WMMA.
// ---------------------------------------------------------------------------
struct DecArgs {
  const bf16 *Ac, *Ap, *insE, *hisE;   // [B*64,512] [B*128,1024] [B*64,512] [B*128,512]
  const float *Hbase, *c0, *h0, *bdec; // Hbase [LACT*B,512]
  const bf16 *Wih, *Whh, *Wbd;         // [2048,512] [2048,512] [512,2944]
  float *sc, *sp, *zcf;                // [B*64] [B*128] [B*512]
  bf16 *zcb, *zpb, *hkb, *outs;        // [B*512]x3, outs [B*LACT,512]
};

__global__ void k_decoder(DecArgs A) {
  extern __shared__ bf16 hbuf[];                 // 2 * 65536 bf16
  const int tid = threadIdx.x;
  const int wave = tid >> 5, l = tid & 31, n = l & 15, mb = (l >> 4) * 8;

  for (int i = tid; i < B_ * ACT_H_; i += blockDim.x) hbuf[i] = (bf16)A.h0[i];
  v8f carr[8];
  {
    int idx = 0;
    for (int gb = wave; gb < 256; gb += 32, ++idx) {
      const int mt = gb >> 5, nt = gb & 31;
#pragma unroll
      for (int e = 0; e < 8; ++e)
        carr[idx][e] = A.c0[(size_t)(mt * 16 + mb + e) * ACT_H_ + nt * 16 + n];
    }
  }
  __syncthreads();

  int cur = 0;
  for (int t = 0; t < LACT_; ++t) {
    const bf16* h = hbuf + cur * 65536;
    bf16* hnext = hbuf + (cur ^ 1) * 65536;

    // P1: s_c[b][l] = A_c[b][l] . h[b]
    for (int p = tid; p < B_ * LINS_; p += blockDim.x) {
      const int b = p >> 6;
      const bf16* arow = A.Ac + (size_t)p * 512;
      const bf16* hrow = h + b * 512;
      float s = 0.f;
#pragma unroll 8
      for (int k = 0; k < 512; ++k) s += (float)arow[k] * (float)hrow[k];
      A.sc[p] = s;
    }
    __threadfence_block(); __syncthreads();
    // P2: softmax over l=64
    if (tid < B_) {
      float mx = -1e30f;
#pragma unroll 8
      for (int li = 0; li < LINS_; ++li) mx = fmaxf(mx, A.sc[tid * LINS_ + li]);
      float sm = 0.f;
#pragma unroll 8
      for (int li = 0; li < LINS_; ++li) {
        const float e = __expf(A.sc[tid * LINS_ + li] - mx);
        A.sc[tid * LINS_ + li] = e; sm += e;
      }
      const float inv = 1.f / sm;
#pragma unroll 8
      for (int li = 0; li < LINS_; ++li) A.sc[tid * LINS_ + li] *= inv;
    }
    __threadfence_block(); __syncthreads();
    // P3: z_c[b][d]
    for (int p = tid; p < B_ * 512; p += blockDim.x) {
      const int b = p >> 9, d = p & 511;
      float s = 0.f;
#pragma unroll 8
      for (int li = 0; li < LINS_; ++li)
        s += A.sc[b * LINS_ + li] * (float)A.insE[((size_t)(b * LINS_ + li)) * 512 + d];
      A.zcf[p] = s; A.zcb[p] = (bf16)s;
    }
    __threadfence_block(); __syncthreads();
    // P4: s_p[b][l] = A_p[b][l] . [h, z_c]
    for (int p = tid; p < B_ * LHIS_; p += blockDim.x) {
      const int b = p >> 7;
      const bf16* arow = A.Ap + (size_t)p * 1024;
      const bf16* hrow = h + b * 512;
      const float* zrow = A.zcf + b * 512;
      float s = 0.f;
#pragma unroll 8
      for (int k = 0; k < 512; ++k) s += (float)arow[k] * (float)hrow[k];
#pragma unroll 8
      for (int k = 0; k < 512; ++k) s += (float)arow[512 + k] * zrow[k];
      A.sp[p] = s;
    }
    __threadfence_block(); __syncthreads();
    // P5: softmax over l=128
    if (tid < B_) {
      float mx = -1e30f;
#pragma unroll 8
      for (int li = 0; li < LHIS_; ++li) mx = fmaxf(mx, A.sp[tid * LHIS_ + li]);
      float sm = 0.f;
#pragma unroll 8
      for (int li = 0; li < LHIS_; ++li) {
        const float e = __expf(A.sp[tid * LHIS_ + li] - mx);
        A.sp[tid * LHIS_ + li] = e; sm += e;
      }
      const float inv = 1.f / sm;
#pragma unroll 8
      for (int li = 0; li < LHIS_; ++li) A.sp[tid * LHIS_ + li] *= inv;
    }
    __threadfence_block(); __syncthreads();
    // P6: z_p[b][d]
    for (int p = tid; p < B_ * 512; p += blockDim.x) {
      const int b = p >> 9, d = p & 511;
      float s = 0.f;
#pragma unroll 8
      for (int li = 0; li < LHIS_; ++li)
        s += A.sp[b * LHIS_ + li] * (float)A.hisE[((size_t)(b * LHIS_ + li)) * 512 + d];
      A.zpb[p] = (bf16)s;
    }
    __threadfence_block(); __syncthreads();
    // P7: hk = tanh(Hbase[t] + z_c@Wbd[:, :512]^T + z_p@Wbd[:, 512:1024]^T)
    for (int tile = wave; tile < 8 * 32; tile += 32) {
      const int mt = tile >> 5, nt = tile & 31;
      v8f acc;
#pragma unroll
      for (int e = 0; e < 8; ++e)
        acc[e] = A.Hbase[(size_t)(t * B_ + mt * 16 + mb + e) * 512 + nt * 16 + n];
      const bf16* wr = A.Wbd + (size_t)(nt * 16) * 2944;
      for (int kb = 0; kb < 512; kb += 32)
        acc = wmma_bf(frag_ld(A.zcb + (size_t)mt * 16 * 512 + kb, 512),
                      frag_ld(wr + kb, 2944), acc);
      for (int kb = 0; kb < 512; kb += 32)
        acc = wmma_bf(frag_ld(A.zpb + (size_t)mt * 16 * 512 + kb, 512),
                      frag_ld(wr + 512 + kb, 2944), acc);
#pragma unroll
      for (int e = 0; e < 8; ++e)
        A.hkb[(size_t)(mt * 16 + mb + e) * 512 + nt * 16 + n] = (bf16)tanhf(acc[e]);
    }
    __threadfence_block(); __syncthreads();
    // P8: decoder LSTM gates = hk@Wih^T + h@Whh^T + b  (WMMA), then cell update
    {
      int idx = 0;
      for (int gb = wave; gb < 256; gb += 32, ++idx) {
        const int mt = gb >> 5, nt = gb & 31;
        v8f g[4];
#pragma unroll
        for (int gi = 0; gi < 4; ++gi) {
          const int ngl = gi * 512 + nt * 16;
          const float bv = A.bdec[ngl + n];
          v8f acc;
#pragma unroll
          for (int e = 0; e < 8; ++e) acc[e] = bv;
          const bf16* wi = A.Wih + (size_t)ngl * 512;
          const bf16* wh = A.Whh + (size_t)ngl * 512;
          for (int kb = 0; kb < 512; kb += 32)
            acc = wmma_bf(frag_ld(A.hkb + (size_t)mt * 16 * 512 + kb, 512),
                          frag_ld(wi + kb, 512), acc);
          for (int kb = 0; kb < 512; kb += 32)
            acc = wmma_bf(frag_ld(h + (size_t)mt * 16 * 512 + kb, 512),
                          frag_ld(wh + kb, 512), acc);
          g[gi] = acc;
        }
#pragma unroll
        for (int e = 0; e < 8; ++e) {
          const float iv = sigf(g[0][e]), fv = sigf(g[1][e]);
          const float gv = tanhf(g[2][e]), ov = sigf(g[3][e]);
          const float cv = fv * carr[idx][e] + iv * gv;
          carr[idx][e] = cv;
          const float hv = ov * tanhf(cv);
          const int brow = mt * 16 + mb + e, col = nt * 16 + n;
          hnext[brow * 512 + col] = (bf16)hv;
          A.outs[(size_t)(brow * LACT_ + t) * 512 + col] = (bf16)hv;
        }
      }
    }
    __threadfence_block(); __syncthreads();
    cur ^= 1;
  }
}

// ---------------------------------------------------------------------------
// Host orchestration
// ---------------------------------------------------------------------------
extern "C" void kernel_launch(void* const* d_in, const int* in_sizes, int n_in,
                              void* d_out, int out_size, void* d_ws, size_t ws_size,
                              hipStream_t stream) {
  (void)in_sizes; (void)n_in; (void)out_size; (void)ws_size;
  // inputs
  const int*   ins_tokens = (const int*)d_in[0];
  const int*   his_tokens = (const int*)d_in[1];
  const int*   actions    = (const int*)d_in[2];
  const int*   cur_env    = (const int*)d_in[3];
  const int*   ini_env    = (const int*)d_in[4];
  const float* enc_emb    = (const float*)d_in[5];
  const float* enc_Wih_f  = (const float*)d_in[6];
  const float* enc_Whh_f  = (const float*)d_in[7];
  const float* enc_bih_f  = (const float*)d_in[8];
  const float* enc_bhh_f  = (const float*)d_in[9];
  const float* enc_Wih_b  = (const float*)d_in[10];
  const float* enc_Whh_b  = (const float*)d_in[11];
  const float* enc_bih_b  = (const float*)d_in[12];
  const float* enc_bhh_b  = (const float*)d_in[13];
  const float* color_emb  = (const float*)d_in[14];
  const float* pos_emb    = (const float*)d_in[15];
  const float* ws_Wih     = (const float*)d_in[16];
  const float* ws_Whh     = (const float*)d_in[17];
  const float* ws_bih     = (const float*)d_in[18];
  const float* ws_bhh     = (const float*)d_in[19];
  const float* act_emb    = (const float*)d_in[20];
  const float* W_c        = (const float*)d_in[21];
  const float* W_p        = (const float*)d_in[22];
  // d_in[23..26] (W_sb1/W_sb2/W_sc1/W_sc2) eliminated: broadcast attends are identity
  const float* Wbd_W      = (const float*)d_in[27];
  const float* Wbd_b      = (const float*)d_in[28];
  const float* dec_Wih    = (const float*)d_in[29];
  const float* dec_Whh    = (const float*)d_in[30];
  const float* dec_bih    = (const float*)d_in[31];
  const float* dec_bhh    = (const float*)d_in[32];
  const float* h2a_W      = (const float*)d_in[33];
  const float* h2a_b      = (const float*)d_in[34];
  const float* h0         = (const float*)d_in[35];
  const float* c0         = (const float*)d_in[36];
  float* out = (float*)d_out;

  // workspace carve-out
  size_t off = 0;
  auto alloc = [&](size_t bytes) -> void* {
    void* p = (char*)d_ws + off;
    off += (bytes + 255) & ~(size_t)255;
    return p;
  };
  // bf16 weights
  bf16* WcombF = (bf16*)alloc((size_t)1024 * 384 * 2);
  bf16* WcombB = (bf16*)alloc((size_t)1024 * 384 * 2);
  bf16* wsWcmb = (bf16*)alloc((size_t)128 * 64 * 2);
  bf16* WcB    = (bf16*)alloc((size_t)512 * 512 * 2);
  bf16* WpB    = (bf16*)alloc((size_t)1024 * 512 * 2);
  bf16* WbdB   = (bf16*)alloc((size_t)512 * 2944 * 2);
  bf16* WihB   = (bf16*)alloc((size_t)2048 * 512 * 2);
  bf16* WhhB   = (bf16*)alloc((size_t)2048 * 512 * 2);
  bf16* h2aB   = (bf16*)alloc((size_t)112 * 512 * 2);
  float* bEncF = (float*)alloc(1024 * 4);
  float* bEncB = (float*)alloc(1024 * 4);
  float* bWs   = (float*)alloc(128 * 4);
  float* bDec  = (float*)alloc(2048 * 4);
  float* h2abP = (float*)alloc(112 * 4);
  // activations
  bf16* insX   = (bf16*)alloc((size_t)LINS_ * B_ * INS_E_ * 2);
  bf16* hisX   = (bf16*)alloc((size_t)LHIS_ * B_ * INS_E_ * 2);
  bf16* envCur = (bf16*)alloc((size_t)4 * 896 * 32 * 2);
  bf16* envIni = (bf16*)alloc((size_t)4 * 896 * 32 * 2);
  bf16* actX   = (bf16*)alloc((size_t)LACT_ * B_ * ACT_E_ * 2);
  bf16* insEnc = (bf16*)alloc((size_t)B_ * LINS_ * 512 * 2);
  bf16* hisEnc = (bf16*)alloc((size_t)B_ * LHIS_ * 512 * 2);
  float* envHsC = (float*)alloc((size_t)4 * 896 * 32 * 4);
  float* envHsI = (float*)alloc((size_t)4 * 896 * 32 * 4);
  bf16* envcat = (bf16*)alloc((size_t)B_ * 1792 * 2);
  bf16* AcB    = (bf16*)alloc((size_t)B_ * LINS_ * 512 * 2);
  bf16* ApB    = (bf16*)alloc((size_t)B_ * LHIS_ * 1024 * 2);
  float* C0v   = (float*)alloc((size_t)B_ * 512 * 4);
  float* Hbase = (float*)alloc((size_t)LACT_ * B_ * 512 * 4);
  float* sc    = (float*)alloc((size_t)B_ * LINS_ * 4);
  float* sp    = (float*)alloc((size_t)B_ * LHIS_ * 4);
  float* zcf   = (float*)alloc((size_t)B_ * 512 * 4);
  bf16* zcb    = (bf16*)alloc((size_t)B_ * 512 * 2);
  bf16* zpb    = (bf16*)alloc((size_t)B_ * 512 * 2);
  bf16* hkb    = (bf16*)alloc((size_t)B_ * 512 * 2);
  bf16* outsB  = (bf16*)alloc((size_t)B_ * LACT_ * 512 * 2);

  auto cdiv = [](int a, int b) { return (a + b - 1) / b; };

  // --- weight conversion (fused [Wih|Whh] per LSTM run) ---
  k_conv<<<cdiv(1024 * 128, 256), 256, 0, stream>>>(enc_Wih_f, WcombF, 1024, 128, 384, 0);
  k_conv<<<cdiv(1024 * 256, 256), 256, 0, stream>>>(enc_Whh_f, WcombF, 1024, 256, 384, 128);
  k_conv<<<cdiv(1024 * 128, 256), 256, 0, stream>>>(enc_Wih_b, WcombB, 1024, 128, 384, 0);
  k_conv<<<cdiv(1024 * 256, 256), 256, 0, stream>>>(enc_Whh_b, WcombB, 1024, 256, 384, 128);
  k_conv<<<cdiv(128 * 32, 256), 256, 0, stream>>>(ws_Wih, wsWcmb, 128, 32, 64, 0);
  k_conv<<<cdiv(128 * 32, 256), 256, 0, stream>>>(ws_Whh, wsWcmb, 128, 32, 64, 32);
  k_conv<<<cdiv(2048 * 512, 256), 256, 0, stream>>>(dec_Wih, WihB, 2048, 512, 512, 0);
  k_conv<<<cdiv(2048 * 512, 256), 256, 0, stream>>>(dec_Whh, WhhB, 2048, 512, 512, 0);
  k_conv<<<cdiv(512 * 2944, 256), 256, 0, stream>>>(Wbd_W, WbdB, 512, 2944, 2944, 0);
  k_convT<<<cdiv(512 * 512, 256), 256, 0, stream>>>(W_c, WcB, 512, 512);
  k_convT<<<cdiv(512 * 1024, 256), 256, 0, stream>>>(W_p, WpB, 512, 1024);
  k_pad_h2a<<<cdiv(112 * 512, 256), 256, 0, stream>>>(h2a_W, h2a_b, h2aB, h2abP);
  k_addv<<<cdiv(1024, 256), 256, 0, stream>>>(enc_bih_f, enc_bhh_f, bEncF, 1024);
  k_addv<<<cdiv(1024, 256), 256, 0, stream>>>(enc_bih_b, enc_bhh_b, bEncB, 1024);
  k_addv<<<1, 128, 0, stream>>>(ws_bih, ws_bhh, bWs, 128);
  k_addv<<<cdiv(2048, 256), 256, 0, stream>>>(dec_bih, dec_bhh, bDec, 2048);

  // --- embedding gathers ---
  k_gather_tok<<<cdiv(B_ * LINS_ * INS_E_, 256), 256, 0, stream>>>(ins_tokens, enc_emb, insX, B_, LINS_);
  k_gather_tok<<<cdiv(B_ * LHIS_ * INS_E_, 256), 256, 0, stream>>>(his_tokens, enc_emb, hisX, B_, LHIS_);
  k_gather_env<<<cdiv(4 * 896 * 32, 256), 256, 0, stream>>>(cur_env, color_emb, envCur);
  k_gather_env<<<cdiv(4 * 896 * 32, 256), 256, 0, stream>>>(ini_env, color_emb, envIni);
  k_gather_act<<<cdiv(LACT_ * B_ * ACT_E_, 256), 256, 0, stream>>>(actions, act_emb, actX);

  // --- all 6 LSTM scans in one launch (one persistent block each) ---
  Lstm6 L;
  L.c[0] = {insX, WcombF, bEncF, nullptr, insEnc, LINS_, B_, INS_H_, INS_E_, LINS_ * 512, 512, 0, 0};
  L.c[1] = {insX, WcombB, bEncB, nullptr, insEnc, LINS_, B_, INS_H_, INS_E_, LINS_ * 512, 512, 256, 1};
  L.c[2] = {hisX, WcombF, bEncF, nullptr, hisEnc, LHIS_, B_, INS_H_, INS_E_, LHIS_ * 512, 512, 0, 0};
  L.c[3] = {hisX, WcombB, bEncB, nullptr, hisEnc, LHIS_, B_, INS_H_, INS_E_, LHIS_ * 512, 512, 256, 1};
  L.c[4] = {envCur, wsWcmb, bWs, envHsC, nullptr, 4, 896, 32, 32, 32, 896 * 32, 0, 0};
  L.c[5] = {envIni, wsWcmb, bWs, envHsI, nullptr, 4, 896, 32, 32, 32, 896 * 32, 0, 0};
  k_lstm<<<6, 1024, 2 * 32768 * sizeof(bf16), stream>>>(L);

  // --- env context -> [ini,ini,cur,cur] (broadcast attends are identity) ---
  k_ctx<<<cdiv(B_ * ENVD_, 256), 256, 0, stream>>>(envHsI + 3 * 896 * 32, pos_emb, envcat, 0, 448);
  k_ctx<<<cdiv(B_ * ENVD_, 256), 256, 0, stream>>>(envHsC + 3 * 896 * 32, pos_emb, envcat, 896, 1344);

  auto gemm = [&](const bf16* Aa, int lda, const bf16* Ww, int ldw, const float* bias,
                  const float* cinit, int cmod, int cld, float* outf, bf16* outbf,
                  int old_, int M, int Npad, int Nreal, int K) {
    const int tiles = (M / 16) * (Npad / 16);
    k_gemm<<<cdiv(tiles, 8), 256, 0, stream>>>(Aa, lda, Ww, ldw, bias, cinit, cmod, cld,
                                               outf, outbf, old_, M, Npad, Nreal, K);
  };

  // --- attention projections ---
  gemm(insEnc, 512, WcB, 512, nullptr, nullptr, 0, 0, nullptr, AcB, 512,
       B_ * LINS_, 512, 512, 512);
  gemm(hisEnc, 512, WpB, 512, nullptr, nullptr, 0, 0, nullptr, ApB, 1024,
       B_ * LHIS_, 1024, 1024, 512);
  // --- step-invariant part of hk1 @ Wbd^T: env chunk + bias, then + x_t chunk ---
  gemm(envcat, 1792, WbdB + 1024, 2944, Wbd_b, nullptr, 0, 0, C0v, nullptr, 512,
       B_, 512, 512, 1792);
  gemm(actX, 128, WbdB + 2816, 2944, nullptr, C0v, B_, 512, Hbase, nullptr, 512,
       LACT_ * B_, 512, 512, 128);

  // --- decoder scan ---
  DecArgs D;
  D.Ac = AcB; D.Ap = ApB; D.insE = insEnc; D.hisE = hisEnc;
  D.Hbase = Hbase; D.c0 = c0; D.h0 = h0; D.bdec = bDec;
  D.Wih = WihB; D.Whh = WhhB; D.Wbd = WbdB;
  D.sc = sc; D.sp = sp; D.zcf = zcf;
  D.zcb = zcb; D.zpb = zpb; D.hkb = hkb; D.outs = outsB;
  k_decoder<<<1, 1024, 2 * 65536 * sizeof(bf16), stream>>>(D);

  // --- logits: [B*LACT,512] @ h2a^T (N padded 100 -> 112, store-guarded) ---
  gemm(outsB, 512, h2aB, 512, h2abP, nullptr, 0, 0, out, nullptr, ACTSZ_,
       B_ * LACT_, 112, ACTSZ_, 512);
}
